// Actor_18425409700301
// MI455X (gfx1250) — matrix operand order repaired and checked by reading, not compiled, not run on hardware
//
#include <hip/hip_runtime.h>
#include <hip/hip_bf16.h>

typedef _Float16 v16h __attribute__((ext_vector_type(16)));
typedef _Float16 v8h  __attribute__((ext_vector_type(8)));
typedef float    v8f  __attribute__((ext_vector_type(8)));

#define STATE_D 17
#define H1      400
#define K1P     32      // layer-1 K padded 17 -> 32
#define NT1     25      // 400 / 16
#define H2      300
#define H2P     304     // layer-2 N padded 300 -> 304
#define K2P     416     // layer-2 K padded 400 -> 416
#define NT2     19      // 304 / 16
#define KC2     13      // 416 / 32
#define K3P     320     // layer-3 K padded 300 -> 320
#define KC3     10      // 320 / 32
#define AD      6
#define WAVES   2
#define ROWS_PER_BLOCK (WAVES * 16)

// ---------------------------------------------------------------------------
// Kernel 0: convert + pad weights to f16 in d_ws.
//   W1h: [400][32]  f16  (col >= 17 zero)
//   W2h: [304][416] f16  (row >= 300 or col >= 400 zero)
//   W3h: [16][320]  f16  (row >= 6  or col >= 300 zero)
// ---------------------------------------------------------------------------
__global__ void prep_weights(const float* __restrict__ l1_w,
                             const float* __restrict__ l2_w,
                             const float* __restrict__ l3_w,
                             _Float16* __restrict__ W1h,
                             _Float16* __restrict__ W2h,
                             _Float16* __restrict__ W3h) {
    const int N1 = H1 * K1P;
    const int N2 = H2P * K2P;
    const int N3 = 16 * K3P;
    int idx = blockIdx.x * blockDim.x + threadIdx.x;
    if (idx < N1) {
        int row = idx >> 5, col = idx & 31;
        float v = (col < STATE_D) ? l1_w[row * STATE_D + col] : 0.0f;
        W1h[idx] = (_Float16)v;
    } else if (idx < N1 + N2) {
        int i = idx - N1;
        int row = i / K2P, col = i - row * K2P;
        float v = (row < H2 && col < H1) ? l2_w[row * H1 + col] : 0.0f;
        W2h[i] = (_Float16)v;
    } else if (idx < N1 + N2 + N3) {
        int i = idx - N1 - N2;
        int row = i / K3P, col = i - row * K3P;
        float v = (row < AD && col < H2) ? l3_w[row * H2 + col] : 0.0f;
        W3h[i] = (_Float16)v;
    }
}

// ---------------------------------------------------------------------------
// Main kernel: 2 waves per block, 16 rows per wave. All three layers on WMMA.
// ---------------------------------------------------------------------------
__global__ __launch_bounds__(WAVES * 32)
void actor_kernel(const float* __restrict__ state,
                  const float* __restrict__ b1,
                  const float* __restrict__ b2,
                  const float* __restrict__ b3,
                  const _Float16* __restrict__ W1h,
                  const _Float16* __restrict__ W2h,
                  const _Float16* __restrict__ W3h,
                  float* __restrict__ out,
                  int nrows) {
    __shared__ __align__(16) _Float16 h1[WAVES][16][K2P];   // 26,624 B
    __shared__ __align__(16) _Float16 h2[WAVES][16][K3P];   // 20,480 B
    __shared__ float zbuf[WAVES][16][16];                   //  2,048 B

    const int tid  = threadIdx.x;
    const int wv   = tid >> 5;
    const int lane = tid & 31;
    const int lh   = lane & 15;   // N column / M row within tile
    const int hi   = lane >> 4;   // which K half-group this lane serves
    const int row0 = (blockIdx.x * WAVES + wv) * 16;
    int grow = row0 + lh;
    if (grow >= nrows) grow = nrows - 1;          // clamp (keeps EXEC uniform)
    const bool live = (row0 + lh) < nrows;

    // ---- Build A1 fragment (16x32 f16, K padded) ----
    const float* srow = state + (long)grow * STATE_D;
    v16h a1;
#pragma unroll
    for (int j = 0; j < 8; ++j) a1[j] = (_Float16)srow[hi * 8 + j];
    a1[8] = hi ? (_Float16)0.0f : (_Float16)srow[16];
#pragma unroll
    for (int j = 9; j < 16; ++j) a1[j] = (_Float16)0.0f;

    // Zero h1 pad columns 400..415 and h2 pad columns 300..319
#pragma unroll
    for (int j = 0; j < 8; ++j)
        h1[wv][lh][H1 + hi * 8 + j] = (_Float16)0.0f;
#pragma unroll
    for (int j = 0; j < 10; ++j)
        h2[wv][lh][H2 + hi * 10 + j] = (_Float16)0.0f;

    // ---- Layer 1: h1 = relu(state @ W1^T + b1), 25 N-tiles x 1 K-chunk ----
    for (int nt = 0; nt < NT1; ++nt) {
        float bias = b1[nt * 16 + lh];
        v8f c;
#pragma unroll
        for (int r = 0; r < 8; ++r) c[r] = bias;
        const _Float16* bp = W1h + (nt * 16 + lh) * K1P + hi * 8;
        v8h blo = *(const v8h*)bp;
        v8h bh8 = *(const v8h*)(bp + 16);
        v16h bfrag;
#pragma unroll
        for (int j = 0; j < 8; ++j) { bfrag[j] = blo[j]; bfrag[8 + j] = bh8[j]; }
        c = __builtin_amdgcn_wmma_f32_16x16x32_f16(false, a1, false, bfrag,
                                                   (short)0, c, false, false);
#pragma unroll
        for (int r = 0; r < 8; ++r)
            h1[wv][r + hi * 8][nt * 16 + lh] = (_Float16)fmaxf(c[r], 0.0f);
    }
    __syncthreads();

    // ---- Build 13 A2 fragments from LDS h1 ----
    v16h a2[KC2];
#pragma unroll
    for (int kc = 0; kc < KC2; ++kc) {
        const _Float16* p = &h1[wv][lh][kc * 32 + hi * 8];
        v8h lo  = *(const v8h*)p;
        v8h hi8 = *(const v8h*)(p + 16);
#pragma unroll
        for (int j = 0; j < 8; ++j) { a2[kc][j] = lo[j]; a2[kc][8 + j] = hi8[j]; }
    }

    // ---- Layer 2: h2 = relu(h1 @ W2^T + b2), 19 N-tiles x 13 K-chunks ----
    for (int nt = 0; nt < NT2; ++nt) {
        int ncol = nt * 16 + lh;
        float bias = (ncol < H2) ? b2[ncol] : 0.0f;
        v8f c;
#pragma unroll
        for (int r = 0; r < 8; ++r) c[r] = bias;
        const _Float16* wrow = W2h + (long)ncol * K2P + hi * 8;
#pragma unroll
        for (int kc = 0; kc < KC2; ++kc) {
            v8h blo = *(const v8h*)(wrow + kc * 32);
            v8h bh8 = *(const v8h*)(wrow + kc * 32 + 16);
            v16h bfrag;
#pragma unroll
            for (int j = 0; j < 8; ++j) { bfrag[j] = blo[j]; bfrag[8 + j] = bh8[j]; }
            c = __builtin_amdgcn_wmma_f32_16x16x32_f16(false, a2[kc], false, bfrag,
                                                       (short)0, c, false, false);
        }
#pragma unroll
        for (int r = 0; r < 8; ++r)
            h2[wv][r + hi * 8][ncol] = (_Float16)fmaxf(c[r], 0.0f);
    }
    __syncthreads();

    // ---- Layer 3 on WMMA: z = h2 @ W3^T + b3 (N padded 6 -> 16) ----
    {
        float bias = (lh < AD) ? b3[lh] : 0.0f;
        v8f c;
#pragma unroll
        for (int r = 0; r < 8; ++r) c[r] = bias;
        const _Float16* wrow = W3h + lh * K3P + hi * 8;
#pragma unroll
        for (int kc = 0; kc < KC3; ++kc) {
            const _Float16* p = &h2[wv][lh][kc * 32 + hi * 8];
            v8h alo = *(const v8h*)p;
            v8h ahi = *(const v8h*)(p + 16);
            v16h afrag;
            v8h blo = *(const v8h*)(wrow + kc * 32);
            v8h bh8 = *(const v8h*)(wrow + kc * 32 + 16);
            v16h bfrag;
#pragma unroll
            for (int j = 0; j < 8; ++j) {
                afrag[j] = alo[j];  afrag[8 + j] = ahi[j];
                bfrag[j] = blo[j];  bfrag[8 + j] = bh8[j];
            }
            c = __builtin_amdgcn_wmma_f32_16x16x32_f16(false, afrag, false, bfrag,
                                                       (short)0, c, false, false);
        }
#pragma unroll
        for (int r = 0; r < 8; ++r)
            zbuf[wv][r + hi * 8][lh] = c[r];
    }
    __syncthreads();

    if (!hi && live) {
        // ---- tanh + exact weighted-L1-ball projection (c = 20) ----
        float q[AD], w[AD], aq[AD], wa[AD];
        float S = 0.0f;
#pragma unroll
        for (int j = 0; j < AD; ++j) {
            q[j]  = tanhf(zbuf[wv][lh][j]);
            w[j]  = fabsf(srow[11 + j]);
            aq[j] = fabsf(q[j]);
            wa[j] = w[j] * aq[j];
            S += wa[j];
        }
        float res[AD];
        if (S <= 20.0f) {
#pragma unroll
            for (int j = 0; j < AD; ++j) res[j] = q[j];
        } else {
            float r[AD], w2[AD];
#pragma unroll
            for (int j = 0; j < AD; ++j) {
                r[j]  = (w[j] > 0.0f) ? aq[j] / fmaxf(w[j], 1e-30f)
                                      : -__builtin_inff();
                w2[j] = w[j] * w[j];
            }
            // 12-comparator sorting network, descending by r (static indices)
#define CSWAP(i, jj)                                                       \
            if (r[i] < r[jj]) {                                            \
                float t;                                                   \
                t = r[i];  r[i]  = r[jj];  r[jj]  = t;                     \
                t = wa[i]; wa[i] = wa[jj]; wa[jj] = t;                     \
                t = w2[i]; w2[i] = w2[jj]; w2[jj] = t;                     \
            }
            CSWAP(0, 5) CSWAP(1, 3) CSWAP(2, 4)
            CSWAP(1, 2) CSWAP(3, 4)
            CSWAP(0, 3) CSWAP(2, 5)
            CSWAP(0, 1) CSWAP(2, 3) CSWAP(4, 5)
            CSWAP(1, 2) CSWAP(3, 4)
#undef CSWAP
            float A = 0.0f, Bs = 0.0f, lamk[AD];
            int kcnt = 0;
#pragma unroll
            for (int j = 0; j < AD; ++j) {
                A  += wa[j];
                Bs += w2[j];
                lamk[j] = (A - 20.0f) / fmaxf(Bs, 1e-30f);
                if (r[j] > lamk[j]) kcnt++;
            }
            int idx = kcnt - 1;
            if (idx < 0) idx = 0;
            float lam = lamk[0];
#pragma unroll
            for (int j = 1; j < AD; ++j) if (j == idx) lam = lamk[j];
            lam = fmaxf(lam, 0.0f);
#pragma unroll
            for (int j = 0; j < AD; ++j) {
                float v = fmaxf(aq[j] - lam * w[j], 0.0f);
                res[j] = (q[j] < 0.0f) ? -v : v;
            }
        }
        float* orow = out + (long)grow * AD;
#pragma unroll
        for (int j = 0; j < AD; ++j) orow[j] = res[j];
    }
}

// ---------------------------------------------------------------------------
extern "C" void kernel_launch(void* const* d_in, const int* in_sizes, int n_in,
                              void* d_out, int out_size, void* d_ws, size_t ws_size,
                              hipStream_t stream) {
    (void)n_in; (void)out_size; (void)ws_size;
    const float* state = (const float*)d_in[0];
    const float* l1_w  = (const float*)d_in[1];
    const float* l1_b  = (const float*)d_in[2];
    const float* l2_w  = (const float*)d_in[3];
    const float* l2_b  = (const float*)d_in[4];
    const float* l3_w  = (const float*)d_in[5];
    const float* l3_b  = (const float*)d_in[6];
    float* out = (float*)d_out;

    _Float16* W1h = (_Float16*)d_ws;                 // 400*32   f16
    _Float16* W2h = W1h + H1 * K1P;                  // 304*416  f16
    _Float16* W3h = W2h + H2P * K2P;                 // 16*320   f16

    const int prep_total = H1 * K1P + H2P * K2P + 16 * K3P;
    prep_weights<<<(prep_total + 255) / 256, 256, 0, stream>>>(l1_w, l2_w, l3_w,
                                                               W1h, W2h, W3h);

    const int nrows = in_sizes[0] / STATE_D;
    const int blocks = (nrows + ROWS_PER_BLOCK - 1) / ROWS_PER_BLOCK;
    actor_kernel<<<blocks, WAVES * 32, 0, stream>>>(state, l1_b, l2_b, l3_b,
                                                    W1h, W2h, W3h, out, nrows);
}
